// NNPhotonicMamba_48567490183439
// MI455X (gfx1250) — matrix operand order repaired and checked by reading, not compiled, not run on hardware
//
#include <hip/hip_runtime.h>
#include <hip/hip_bf16.h>
#include <stdint.h>

// ---------------- problem constants (from reference setup_inputs) -----------
#define DM   1024          // d_model
#define DI   2048          // d_inner
#define DS   16            // d_state
#define DC   4             // d_conv
#define DTR  64            // dt_rank
#define BB   2             // batch
#define LL   1024          // seq len
#define NTOK (BB * LL)     // 2048 tokens
#define XDW  (DTR + 2*DS)  // 96 = x_dbl width

typedef __bf16 bf16;
typedef __attribute__((ext_vector_type(16))) __bf16 v16bf;
typedef __attribute__((ext_vector_type(8)))  __bf16 v8bf;
typedef __attribute__((ext_vector_type(8)))  float  v8f;

// ---------------- helpers ---------------------------------------------------
__device__ __forceinline__ bf16 f2bf(float f) {
  uint32_t u = __builtin_bit_cast(uint32_t, f);
  uint32_t r = u + 0x7FFFu + ((u >> 16) & 1u);   // round-to-nearest-even
  uint16_t h = (uint16_t)(r >> 16);
  return __builtin_bit_cast(bf16, h);
}
// monotonic uint key for float ordering (min float == min key)
__device__ __forceinline__ uint32_t fkey(float f) {
  uint32_t u = __builtin_bit_cast(uint32_t, f);
  return (u & 0x80000000u) ? ~u : (u | 0x80000000u);
}
__device__ __forceinline__ float fkey_dec(uint32_t k) {
  uint32_t u = (k & 0x80000000u) ? (k ^ 0x80000000u) : ~k;
  return __builtin_bit_cast(float, u);
}
__device__ __forceinline__ float silu(float v) {
  return v / (1.0f + __expf(-v));
}

// ---------------- elementwise kernels ---------------------------------------
__global__ void k_cast_bf16(const float* __restrict__ src, bf16* __restrict__ dst, size_t n) {
  size_t i = (size_t)blockIdx.x * blockDim.x + threadIdx.x;
  if (i < n) dst[i] = f2bf(src[i]);
}

// strided slice f32 -> bf16 (dt_raw: cols [0,64) of x_dbl, ld=96)
__global__ void k_slice_bf16(const float* __restrict__ src, bf16* __restrict__ dst,
                             int rows, int cols, int ld) {
  int i = blockIdx.x * blockDim.x + threadIdx.x;
  if (i >= rows * cols) return;
  int r = i / cols, c = i - r * cols;
  dst[i] = f2bf(src[(size_t)r * ld + c]);
}

__global__ void k_init_key(uint32_t* key) { *key = 0xFFFFFFFFu; }

// global min over B slice of x_dbl: cols [64,80), ld = 96
__global__ void k_minB(const float* __restrict__ xdbl, uint32_t* __restrict__ key) {
  __shared__ float sm[256];
  int i = blockIdx.x * blockDim.x + threadIdx.x;   // over NTOK*DS
  float v = 3.4e38f;
  if (i < NTOK * DS) {
    int tok = i >> 4, n = i & 15;
    v = xdbl[(size_t)tok * XDW + DTR + n];
  }
  sm[threadIdx.x] = v; __syncthreads();
  for (int off = 128; off > 0; off >>= 1) {
    if (threadIdx.x < off) sm[threadIdx.x] = fminf(sm[threadIdx.x], sm[threadIdx.x + off]);
    __syncthreads();
  }
  if (threadIdx.x == 0) atomicMin(key, fkey(sm[0]));
}

// delta = clip(softplus(dt + dt_bias), 1e-4, 20) in place
__global__ void k_delta(float* __restrict__ dt, const float* __restrict__ dt_bias, size_t n) {
  size_t i = (size_t)blockIdx.x * blockDim.x + threadIdx.x;
  if (i >= n) return;
  int c = (int)(i % DI);
  float v = dt[i] + dt_bias[c];
  float sp = (v > 20.0f) ? v : log1pf(__expf(v));
  dt[i] = fminf(fmaxf(sp, 1e-4f), 20.0f);
}

__global__ void k_expA(const float* __restrict__ A_log, float* __restrict__ Aexp, int n) {
  int i = blockIdx.x * blockDim.x + threadIdx.x;
  if (i < n) Aexp[i] = __expf(A_log[i]);
}

// causal depthwise conv (k=4) + bias + SiLU; reads x half of xz; writes f32+bf16
__global__ void k_conv_silu(const float* __restrict__ xz, const float* __restrict__ w,
                            const float* __restrict__ bias,
                            float* __restrict__ xs, bf16* __restrict__ xsb) {
  int i = blockIdx.x * blockDim.x + threadIdx.x;       // over NTOK*DI
  if (i >= NTOK * DI) return;
  int c = i % DI;
  int t = (i / DI) % LL;
  int b = i / (DI * LL);
  float acc = bias[c];
  #pragma unroll
  for (int j = 0; j < DC; ++j) {
    int tt = t - (DC - 1) + j;
    if (tt >= 0)
      acc += w[c * DC + j] * xz[(size_t)(b * LL + tt) * (2 * DI) + c];
  }
  float v = silu(acc);
  xs[i]  = v;
  xsb[i] = f2bf(v);
}

// ---------------- selective scan --------------------------------------------
// one thread per (b, channel); 256 threads/block, grid (DI/256, BB)
__global__ void k_scan(const float* __restrict__ delta, const float* __restrict__ xs,
                       const float* __restrict__ xdbl, const float* __restrict__ xz,
                       const float* __restrict__ Aexp, const float* __restrict__ Dv,
                       const uint32_t* __restrict__ betakey,
                       float* __restrict__ y, bf16* __restrict__ yb) {
  __shared__ float sBC[2 * DS];            // B then C for current token
  int b = blockIdx.y;
  int c = blockIdx.x * blockDim.x + threadIdx.x;
  float beta = fmaxf(0.0f, -fkey_dec(*betakey));
  float a[DS], s[DS];
  #pragma unroll
  for (int n = 0; n < DS; ++n) { a[n] = Aexp[c * DS + n]; s[n] = 0.0f; }
  float Dc = Dv[c];
  for (int t = 0; t < LL; ++t) {
    size_t tok = (size_t)(b * LL + t);
    if (threadIdx.x < 2 * DS) sBC[threadIdx.x] = xdbl[tok * XDW + DTR + threadIdx.x];
    __syncthreads();
    float d_ = delta[tok * DI + c];
    float xv = xs[tok * DI + c];
    float yv = 0.0f;
    #pragma unroll
    for (int n = 0; n < DS; ++n) {
      float dA = __expf(-d_ * a[n]);
      s[n] = dA * s[n] + d_ * (sBC[n] + beta) * xv;   // gamma cancels; beta does not
      yv += s[n] * sBC[DS + n];
    }
    yv += xv * Dc;
    float zv = xz[tok * (2 * DI) + DI + c];
    yv *= silu(zv);
    y[tok * DI + c]  = yv;
    yb[tok * DI + c] = f2bf(yv);
    __syncthreads();
  }
}

// ---------------- bf16 WMMA GEMM (simple): one 16x16 tile per wave -----------
// C[M,N] = A[M,K] * B[N,K]^T ; used for the small-N GEMM (N=96).
__global__ void k_wmma_gemm(const bf16* __restrict__ A, const bf16* __restrict__ B,
                            float* __restrict__ C, int M, int N, int K) {
  int wave  = threadIdx.x >> 5;
  int lane  = threadIdx.x & 31;
  int tileN = (blockIdx.x * 8 + wave) * 16;
  int tileM = blockIdx.y * 16;
  if (tileN >= N) return;                 // wave-uniform: EXEC stays all-ones
  int half = lane >> 4;                   // K-half selector
  int l16  = lane & 15;

  const bf16* Arow = A + (size_t)(tileM + l16) * K;             // A: lanes index M
  const bf16* Brow = B + (size_t)(tileN + l16) * K + 16 * half; // B: lanes index N

  v8f acc = {};
  #pragma unroll 2
  for (int k0 = 0; k0 < K; k0 += 32) {
    v16bf a = *(const v16bf*)(Arow + k0 + 16 * half);       // K = k0+16h .. +15
    v16bf bfrag = *(const v16bf*)(Brow + k0);               // 16 contiguous K
    acc = __builtin_amdgcn_wmma_f32_16x16x32_bf16(
        false, a, false, bfrag, (short)0, acc, false, false);
  }
  #pragma unroll
  for (int r = 0; r < 8; ++r)
    C[(size_t)(tileM + r + 8 * half) * N + tileN + l16] = acc[r];
}

// ---------------- bf16 WMMA GEMM (blocked): 32x64 tile per wave --------------
// 8 waves = 2(M) x 4(N); block tile = 64(M) x 256(N); 8 WMMAs per K-step.
// Requires M % 64 == 0, N % 256 == 0, K % 32 == 0.
__global__ void k_wmma_gemm_blk(const bf16* __restrict__ A, const bf16* __restrict__ B,
                                float* __restrict__ C, int M, int N, int K) {
  int wave = threadIdx.x >> 5;
  int lane = threadIdx.x & 31;
  int wm   = wave >> 2;                   // 0..1
  int wn   = wave & 3;                    // 0..3
  int tileM = blockIdx.y * 64 + wm * 32;
  int tileN = blockIdx.x * 256 + wn * 64;
  int half = lane >> 4;
  int l16  = lane & 15;

  const bf16* Arow[2];
  Arow[0] = A + (size_t)(tileM + l16) * K;
  Arow[1] = A + (size_t)(tileM + 16 + l16) * K;
  const bf16* Brow[4];
  #pragma unroll
  for (int j = 0; j < 4; ++j)
    Brow[j] = B + (size_t)(tileN + 16 * j + l16) * K + 16 * half;

  v8f acc[2][4] = {};
  for (int k0 = 0; k0 < K; k0 += 32) {
    // prefetch ahead into the cache hierarchy (global_prefetch_b8)
    if (k0 + 128 < K) {
      __builtin_prefetch(Arow[0] + k0 + 128, 0, 3);
      __builtin_prefetch(Arow[1] + k0 + 128, 0, 3);
      __builtin_prefetch(Brow[0] + k0 + 128, 0, 3);
      __builtin_prefetch(Brow[2] + k0 + 128, 0, 3);
    }
    v16bf a[2];
    #pragma unroll
    for (int i = 0; i < 2; ++i) {
      a[i] = *(const v16bf*)(Arow[i] + k0 + 16 * half);
    }
    #pragma unroll
    for (int j = 0; j < 4; ++j) {
      v16bf bfrag = *(const v16bf*)(Brow[j] + k0);
      #pragma unroll
      for (int i = 0; i < 2; ++i)
        acc[i][j] = __builtin_amdgcn_wmma_f32_16x16x32_bf16(
            false, a[i], false, bfrag, (short)0, acc[i][j], false, false);
    }
  }
  #pragma unroll
  for (int i = 0; i < 2; ++i)
    #pragma unroll
    for (int j = 0; j < 4; ++j)
      #pragma unroll
      for (int r = 0; r < 8; ++r)
        C[(size_t)(tileM + 16 * i + r + 8 * half) * N + tileN + 16 * j + l16] = acc[i][j][r];
}

// ---------------- launcher ---------------------------------------------------
static char* wsal(char*& p, size_t bytes) {
  char* r = p;
  p += (bytes + 255) & ~(size_t)255;
  return r;
}

extern "C" void kernel_launch(void* const* d_in, const int* in_sizes, int n_in,
                              void* d_out, int out_size, void* d_ws, size_t ws_size,
                              hipStream_t stream) {
  const float* hs      = (const float*)d_in[0]; // (2,1024,1024)
  const float* W_in    = (const float*)d_in[1]; // (4096,1024)
  const float* conv_w  = (const float*)d_in[2]; // (2048,1,4)
  const float* conv_b  = (const float*)d_in[3]; // (2048,)
  const float* W_x     = (const float*)d_in[4]; // (96,2048)
  const float* W_dt    = (const float*)d_in[5]; // (2048,64)
  const float* dt_bias = (const float*)d_in[6]; // (2048,)
  const float* W_out   = (const float*)d_in[7]; // (1024,2048)
  const float* A_log   = (const float*)d_in[8]; // (2048,16)
  const float* Dv      = (const float*)d_in[9]; // (2048,)
  float* out = (float*)d_out;                   // (2048,1024) f32

  char* p = (char*)d_ws;
  bf16*  hs_b   = (bf16*)wsal(p, (size_t)NTOK * DM * 2);
  bf16*  Win_b  = (bf16*)wsal(p, (size_t)2 * DI * DM * 2);
  bf16*  Wx_b   = (bf16*)wsal(p, (size_t)XDW * DI * 2);
  bf16*  Wdt_b  = (bf16*)wsal(p, (size_t)DI * DTR * 2);
  bf16*  Wout_b = (bf16*)wsal(p, (size_t)DM * DI * 2);
  float* xz     = (float*)wsal(p, (size_t)NTOK * 2 * DI * 4);
  float* xs     = (float*)wsal(p, (size_t)NTOK * DI * 4);
  bf16*  xs_b   = (bf16*)wsal(p, (size_t)NTOK * DI * 2);
  float* xdbl   = (float*)wsal(p, (size_t)NTOK * XDW * 4);
  bf16*  dtr_b  = (bf16*)wsal(p, (size_t)NTOK * DTR * 2);
  float* dt     = (float*)wsal(p, (size_t)NTOK * DI * 4);
  float* Aexp   = (float*)wsal(p, (size_t)DI * DS * 4);
  float* yv     = (float*)wsal(p, (size_t)NTOK * DI * 4);
  bf16*  y_b    = (bf16*)wsal(p, (size_t)NTOK * DI * 2);
  uint32_t* bkey = (uint32_t*)wsal(p, 4);
  (void)ws_size; (void)in_sizes; (void)n_in; (void)out_size;

  auto grid1 = [](size_t n) { return dim3((unsigned)((n + 255) / 256)); };

  // casts to bf16
  k_cast_bf16<<<grid1((size_t)NTOK * DM), 256, 0, stream>>>(hs,    hs_b,   (size_t)NTOK * DM);
  k_cast_bf16<<<grid1((size_t)2 * DI * DM), 256, 0, stream>>>(W_in, Win_b,  (size_t)2 * DI * DM);
  k_cast_bf16<<<grid1((size_t)XDW * DI), 256, 0, stream>>>(W_x,    Wx_b,   (size_t)XDW * DI);
  k_cast_bf16<<<grid1((size_t)DI * DTR), 256, 0, stream>>>(W_dt,   Wdt_b,  (size_t)DI * DTR);
  k_cast_bf16<<<grid1((size_t)DM * DI), 256, 0, stream>>>(W_out,  Wout_b, (size_t)DM * DI);
  k_init_key<<<1, 1, 0, stream>>>(bkey);
  k_expA<<<grid1((size_t)DI * DS), 256, 0, stream>>>(A_log, Aexp, DI * DS);

  // GEMM1: xz[N,4096] = hs[N,1024] @ W_in^T   (blocked WMMA)
  k_wmma_gemm_blk<<<dim3((2 * DI) / 256, NTOK / 64), 256, 0, stream>>>(hs_b, Win_b, xz, NTOK, 2 * DI, DM);

  // depthwise causal conv + SiLU
  k_conv_silu<<<grid1((size_t)NTOK * DI), 256, 0, stream>>>(xz, conv_w, conv_b, xs, xs_b);

  // GEMM2: x_dbl[N,96] = xs[N,2048] @ W_x^T   (simple WMMA, N=96)
  k_wmma_gemm<<<dim3((XDW + 127) / 128, NTOK / 16), 256, 0, stream>>>(xs_b, Wx_b, xdbl, NTOK, XDW, DI);

  // beta = max(0, -min(Bm)) : reduce over B slice
  k_minB<<<grid1((size_t)NTOK * DS), 256, 0, stream>>>(xdbl, bkey);

  // dt_raw slice -> bf16 ; GEMM3: dt[N,2048] = dt_raw[N,64] @ W_dt^T (blocked)
  k_slice_bf16<<<grid1((size_t)NTOK * DTR), 256, 0, stream>>>(xdbl, dtr_b, NTOK, DTR, XDW);
  k_wmma_gemm_blk<<<dim3(DI / 256, NTOK / 64), 256, 0, stream>>>(dtr_b, Wdt_b, dt, NTOK, DI, DTR);

  // delta = clip(softplus(dt + bias), 1e-4, 20)
  k_delta<<<grid1((size_t)NTOK * DI), 256, 0, stream>>>(dt, dt_bias, (size_t)NTOK * DI);

  // selective scan + D skip + SiLU(z) gate
  k_scan<<<dim3(DI / 256, BB), 256, 0, stream>>>(dt, xs, xdbl, xz, Aexp, Dv, bkey, yv, y_b);

  // GEMM4: out[N,1024] = y[N,2048] @ W_out^T  (blocked)
  k_wmma_gemm_blk<<<dim3(DM / 256, NTOK / 64), 256, 0, stream>>>(y_b, Wout_b, out, NTOK, DM, DI);
}
